// RACDecoder_84421877170740
// MI455X (gfx1250) — matrix operand order repaired and checked by reading, not compiled, not run on hardware
//
#include <hip/hip_runtime.h>
#include <math.h>

// ---------------------------------------------------------------------------
// RAC decoder forward for MI455X (gfx1250, wave32, WMMA bf16 16x16x32).
//
// Sizes (fixed by the reference):
//   B=4, S=2048, D=512, Y=8922, V=50000, E=100, T=32, K=9
// Outputs (concatenated f32): y[B,Y], loss[1], alpha[B,Y,S], m[B,Y,D]
// ---------------------------------------------------------------------------

#define B_    4
#define S_    2048
#define D_    512
#define Y_    8922
#define YPAD  8928          // 558 * 16
#define V_    50000
#define E_    100
#define T_    32
#define KW    9
#define PADW  4             // KW/2
#define KC    900           // E_*KW  (implicit-GEMM K for the conv)
#define KCP   928           // KC padded to multiple of 32
#define PROW  944           // LDS row stride (elements) for the patch matrix

typedef __attribute__((ext_vector_type(8)))  __bf16 bhalf8;
typedef __attribute__((ext_vector_type(16))) __bf16 bhalf16;
typedef __attribute__((ext_vector_type(8)))  float  floatx8;

// --- WMMA helper: D = A(16x32 bf16) * B(32x16 bf16) + C(16x16 f32) ---------
__device__ __forceinline__ floatx8 wmma_bf16(bhalf16 a, bhalf16 b, floatx8 c) {
  return __builtin_amdgcn_wmma_f32_16x16x32_bf16(false, a, false, b,
                                                 (short)0, c, false, false);
}

// A-fragment (16x32, MxK). rowptr = &A[m][k0] with m = lane&15.
// lanes 0-15: K {k0..k0+7, k0+16..k0+23}; lanes 16-31: K {+8..+15, +24..+31}
__device__ __forceinline__ bhalf16 frag_a(const __bf16* rowptr, int half) {
  const __bf16* p = rowptr + half * 8;
  bhalf8 lo = *(const bhalf8*)p;
  bhalf8 hi = *(const bhalf8*)(p + 16);
  return __builtin_shufflevector(lo, hi, 0,1,2,3,4,5,6,7,8,9,10,11,12,13,14,15);
}

// B-fragment (32x16, KxN) from B^T rows. colptr = &Bt[n][k0] with n = lane&15.
// lanes 0-15: K {k0..k0+7, k0+16..k0+23}; lanes 16-31: K {+8..+15, +24..+31}
__device__ __forceinline__ bhalf16 frag_b(const __bf16* colptr, int half) {
  const __bf16* p = colptr + half * 8;
  bhalf8 lo = *(const bhalf8*)p;
  bhalf8 hi = *(const bhalf8*)(p + 16);
  return __builtin_shufflevector(lo, hi, 0,1,2,3,4,5,6,7,8,9,10,11,12,13,14,15);
}

// ---------------------------------------------------------------------------
// Prep: x (f32 [B,S,D]) -> xbf [B,S,D] bf16 and xTbf [B,D,S] bf16
// ---------------------------------------------------------------------------
__global__ __launch_bounds__(256) void convert_x_kernel(const float* __restrict__ x,
                                                        __bf16* __restrict__ xbf,
                                                        __bf16* __restrict__ xTbf) {
  int i = blockIdx.x * 256 + threadIdx.x;
  if (i >= B_ * S_ * D_) return;
  float v = x[i];
  __bf16 bv = (__bf16)v;
  xbf[i] = bv;
  int d = i & (D_ - 1);
  int s = (i >> 9) & (S_ - 1);
  int b = i >> 20;                    // S_*D_ = 2^20
  xTbf[((size_t)b * D_ + d) * S_ + s] = bv;
}

// conv_w [D,E,K] f32 -> Wbf [D][KCP] bf16 with col = k*E + e, zero padded.
__global__ __launch_bounds__(256) void pack_w_kernel(const float* __restrict__ conv_w,
                                                     __bf16* __restrict__ Wbf) {
  int i = blockIdx.x * 256 + threadIdx.x;
  if (i >= D_ * KCP) return;
  int d = i / KCP, col = i - d * KCP;
  float v = 0.f;
  if (col < KC) {
    int k = col / E_, e = col - k * E_;
    v = conv_w[((size_t)d * E_ + e) * KW + k];
  }
  Wbf[i] = (__bf16)v;
}

// ---------------------------------------------------------------------------
// Kernel A: title encoder.  One block per code y.
// Implicit GEMM: P[32 x 928] (patch, LDS bf16)  x  Wbf^T (928 x 512)
// then max over t (the M dim), +bias, tanh -> Ubf[y][0..511]
// ---------------------------------------------------------------------------
__global__ __launch_bounds__(256) void title_conv_kernel(
    const int* __restrict__ c2title, const float* __restrict__ embed_W,
    const float* __restrict__ conv_b, const __bf16* __restrict__ Wbf,
    __bf16* __restrict__ Ubf) {
  int y = blockIdx.x;
  int tid = threadIdx.x;
  if (y >= Y_) {                       // zero the pad rows of Ubf
    Ubf[(size_t)y * D_ + tid * 2]     = (__bf16)0.f;
    Ubf[(size_t)y * D_ + tid * 2 + 1] = (__bf16)0.f;
    return;
  }
  __shared__ __bf16 P[T_ * PROW];
  __shared__ int toks[T_];
  if (tid < T_) toks[tid] = c2title[y * T_ + tid];
  __syncthreads();

  // Build the patch matrix: P[t][k*E+e] = embed_W[tok[t+k-4]][e] (0 at edges)
  for (int idx = tid; idx < T_ * KCP; idx += 256) {
    int t = idx / KCP;
    int col = idx - t * KCP;
    float v = 0.f;
    if (col < KC) {
      int k = col / E_, e = col - k * E_;
      int tt = t + k - PADW;
      if (tt >= 0 && tt < T_) v = embed_W[(size_t)toks[tt] * E_ + e];
    }
    P[t * PROW + col] = (__bf16)v;
  }
  __syncthreads();

  int w = tid >> 5, lane = tid & 31, half = lane >> 4, lm = lane & 15;
  floatx8 z = (floatx8)0.0f;
  floatx8 acc[2][4];                  // [m-tile (t 0-15 / 16-31)][n-tile]
#pragma unroll
  for (int mt = 0; mt < 2; ++mt)
#pragma unroll
    for (int i = 0; i < 4; ++i) acc[mt][i] = z;

  for (int ks = 0; ks < KCP / 32; ++ks) {
    int k0 = ks * 32;
    bhalf16 a0 = frag_a(&P[(0  + lm) * PROW + k0], half);
    bhalf16 a1 = frag_a(&P[(16 + lm) * PROW + k0], half);
#pragma unroll
    for (int i = 0; i < 4; ++i) {
      int d0 = w * 64 + i * 16;
      bhalf16 bf = frag_b(&Wbf[(size_t)(d0 + lm) * KCP + k0], half);
      acc[0][i] = wmma_bf16(a0, bf, acc[0][i]);
      acc[1][i] = wmma_bf16(a1, bf, acc[1][i]);
    }
  }

  // maxpool over t (M dim of both tiles), + bias, tanh
#pragma unroll
  for (int i = 0; i < 4; ++i) {
    float mx = -3.4e38f;
#pragma unroll
    for (int r = 0; r < 8; ++r)
      mx = fmaxf(mx, fmaxf(acc[0][i][r], acc[1][i][r]));
    mx = fmaxf(mx, __shfl_xor(mx, 16));        // combine the two lane halves
    if (half == 0) {
      int d = w * 64 + i * 16 + lm;
      float u = tanhf(mx + conv_b[d]);
      Ubf[(size_t)y * D_ + d] = (__bf16)u;
    }
  }
}

// ---------------------------------------------------------------------------
// Kernel B1: scores + softmax -> alpha (f32).
// Block = 16 labels x full S=2048. 512 threads = 16 waves, 8 s-tiles each
// (64 acc VGPRs/wave -> no accumulator shuffling, deep load pipelining).
// ---------------------------------------------------------------------------
__global__ __launch_bounds__(512) void attn_scores_kernel(
    const __bf16* __restrict__ Ubf, const __bf16* __restrict__ xbf,
    float* __restrict__ alpha_out) {
  int y0 = blockIdx.x * 16;
  int b  = blockIdx.y;
  int tid = threadIdx.x, w = tid >> 5, lane = tid & 31;
  int half = lane >> 4, lm = lane & 15;
  const __bf16* xb = xbf + (size_t)b * S_ * D_;

  floatx8 z = (floatx8)0.0f;
  floatx8 acc[8];
#pragma unroll
  for (int i = 0; i < 8; ++i) acc[i] = z;

  int s_base = w * 128;               // 16 waves x 128 columns = 2048
  for (int ks = 0; ks < D_ / 32; ++ks) {
    int k0 = ks * 32;
    bhalf16 a = frag_a(&Ubf[(size_t)(y0 + lm) * D_ + k0], half);
#pragma unroll
    for (int i = 0; i < 8; ++i) {
      int s0 = s_base + i * 16;
      bhalf16 bf = frag_b(&xb[(size_t)(s0 + lm) * D_ + k0], half);
      acc[i] = wmma_bf16(a, bf, acc[i]);
    }
  }

  const float scale = 0.04419417382415922f;   // 1/sqrt(512)
  __shared__ float red[16][16];
  float gmax[8], ginv[8];

  // row max (softmax over s): tiles -> lanes (within half) -> waves via LDS
#pragma unroll
  for (int r = 0; r < 8; ++r) {
    float mx = -3.4e38f;
#pragma unroll
    for (int i = 0; i < 8; ++i) mx = fmaxf(mx, acc[i][r]);
    for (int off = 1; off < 16; off <<= 1) mx = fmaxf(mx, __shfl_xor(mx, off));
    if (lm == 0) red[w][half * 8 + r] = mx;
  }
  __syncthreads();
#pragma unroll
  for (int r = 0; r < 8; ++r) {
    float mx = red[0][half * 8 + r];
#pragma unroll
    for (int ww = 1; ww < 16; ++ww) mx = fmaxf(mx, red[ww][half * 8 + r]);
    gmax[r] = mx;
  }
  __syncthreads();

  // exp + row sum
#pragma unroll
  for (int r = 0; r < 8; ++r) {
    float sm = 0.f;
#pragma unroll
    for (int i = 0; i < 8; ++i) {
      float e = __expf(scale * acc[i][r] - gmax[r]);
      acc[i][r] = e;
      sm += e;
    }
    for (int off = 1; off < 16; off <<= 1) sm += __shfl_xor(sm, off);
    if (lm == 0) red[w][half * 8 + r] = sm;
  }
  __syncthreads();
#pragma unroll
  for (int r = 0; r < 8; ++r) {
    float sm = 0.f;
#pragma unroll
    for (int ww = 0; ww < 16; ++ww) sm += red[ww][half * 8 + r];
    ginv[r] = 1.f / sm;
  }

  // write alpha [B,Y,S] (f32, dword stores; base is only 4B-aligned)
#pragma unroll
  for (int r = 0; r < 8; ++r) {
    int yy = y0 + half * 8 + r;
    if (yy < Y_) {
      float* rowp = alpha_out + ((size_t)b * Y_ + yy) * S_;
#pragma unroll
      for (int i = 0; i < 8; ++i)
        rowp[s_base + i * 16 + lm] = acc[i][r] * ginv[r];
    }
  }
}

// ---------------------------------------------------------------------------
// Kernel B2: m = alpha * x  (+ fused y = dot(m, final_w) + final_b).
// Block = 16 labels x full D=512. A-fragments built on the fly from f32 alpha.
// ---------------------------------------------------------------------------
__global__ __launch_bounds__(256) void attn_context_kernel(
    const float* __restrict__ alpha, const __bf16* __restrict__ xTbf,
    const float* __restrict__ final_w, const float* __restrict__ final_b,
    float* __restrict__ m_out, float* __restrict__ y_out) {
  int y0 = blockIdx.x * 16;
  int b  = blockIdx.y;
  int tid = threadIdx.x, w = tid >> 5, lane = tid & 31;
  int half = lane >> 4, lm = lane & 15;

  __shared__ float ys[16];
  if (tid < 16) ys[tid] = 0.f;
  __syncthreads();

  const __bf16* xT = xTbf + (size_t)b * D_ * S_;
  int arow = y0 + lm; if (arow >= Y_) arow = Y_ - 1;     // clamp pad rows
  const float* arowp = alpha + ((size_t)b * Y_ + arow) * S_;

  floatx8 z = (floatx8)0.0f;
  floatx8 acc[4];
#pragma unroll
  for (int i = 0; i < 4; ++i) acc[i] = z;

  for (int ks = 0; ks < S_ / 32; ++ks) {
    int k0 = ks * 32;
    bhalf16 a;
    const float* p = arowp + k0 + half * 8;
#pragma unroll
    for (int j = 0; j < 8; ++j) a[j] = (__bf16)p[j];
#pragma unroll
    for (int j = 0; j < 8; ++j) a[8 + j] = (__bf16)p[16 + j];
#pragma unroll
    for (int i = 0; i < 4; ++i) {
      int d0 = w * 64 + i * 16;
      bhalf16 bf = frag_b(&xT[(size_t)(d0 + lm) * S_ + k0], half);
      acc[i] = wmma_bf16(a, bf, acc[i]);
    }
  }

  // store m and accumulate y = sum_d m*final_w
  float part[8];
#pragma unroll
  for (int r = 0; r < 8; ++r) part[r] = 0.f;
#pragma unroll
  for (int i = 0; i < 4; ++i) {
    int d = w * 64 + i * 16 + lm;
#pragma unroll
    for (int r = 0; r < 8; ++r) {
      int yy = y0 + half * 8 + r;
      if (yy < Y_) {
        float mv = acc[i][r];
        m_out[((size_t)b * Y_ + yy) * D_ + d] = mv;
        part[r] += mv * final_w[(size_t)yy * D_ + d];
      }
    }
  }
#pragma unroll
  for (int r = 0; r < 8; ++r) {
    float p = part[r];
    for (int off = 1; off < 16; off <<= 1) p += __shfl_xor(p, off);
    if (lm == 0) atomicAdd(&ys[half * 8 + r], p);
  }
  __syncthreads();
  if (tid < 16) {
    int yy = y0 + tid;
    if (yy < Y_) y_out[(size_t)b * Y_ + yy] = ys[tid] + final_b[yy];
  }
}

// ---------------------------------------------------------------------------
// Loss: BCEWithLogits mean over [B,Y]. Single block => deterministic.
// ---------------------------------------------------------------------------
__global__ __launch_bounds__(256) void loss_kernel(const float* __restrict__ y_out,
                                                   const float* __restrict__ target,
                                                   float* __restrict__ loss_out) {
  __shared__ float s[256];
  float acc = 0.f;
  for (int i = threadIdx.x; i < B_ * Y_; i += 256) {
    float yv = y_out[i], t = target[i];
    acc += fmaxf(yv, 0.f) - yv * t + log1pf(__expf(-fabsf(yv)));
  }
  s[threadIdx.x] = acc;
  __syncthreads();
  for (int off = 128; off > 0; off >>= 1) {
    if (threadIdx.x < off) s[threadIdx.x] += s[threadIdx.x + off];
    __syncthreads();
  }
  if (threadIdx.x == 0) loss_out[0] = s[0] / (float)(B_ * Y_);
}

// ---------------------------------------------------------------------------
extern "C" void kernel_launch(void* const* d_in, const int* in_sizes, int n_in,
                              void* d_out, int out_size, void* d_ws, size_t ws_size,
                              hipStream_t stream) {
  const float* x       = (const float*)d_in[0];
  const float* target  = (const float*)d_in[1];
  const int*   c2title = (const int*)  d_in[2];
  const float* embed_W = (const float*)d_in[3];
  const float* conv_w  = (const float*)d_in[4];
  const float* conv_b  = (const float*)d_in[5];
  const float* final_w = (const float*)d_in[6];
  const float* final_b = (const float*)d_in[7];

  // workspace layout (all 16B aligned): ~25.6 MB total
  char* ws = (char*)d_ws;
  __bf16* Wbf  = (__bf16*)(ws);                                   //   950,272 B
  __bf16* Ubf  = (__bf16*)(ws + 950272);                          // 9,142,272 B
  __bf16* xbf  = (__bf16*)(ws + 950272 + 9142272);                // 8,388,608 B
  __bf16* xTbf = (__bf16*)(ws + 950272 + 9142272 + 8388608);      // 8,388,608 B

  float* out       = (float*)d_out;
  float* y_out     = out;                                         // B*Y = 35688
  float* loss_out  = out + 35688;                                 // 1
  float* alpha_out = out + 35689;                                 // B*Y*S
  float* m_out     = out + (size_t)35689 + (size_t)B_ * Y_ * S_;  // B*Y*D

  convert_x_kernel<<<(B_ * S_ * D_ + 255) / 256, 256, 0, stream>>>(x, xbf, xTbf);
  pack_w_kernel<<<(D_ * KCP + 255) / 256, 256, 0, stream>>>(conv_w, Wbf);
  title_conv_kernel<<<YPAD, 256, 0, stream>>>(c2title, embed_W, conv_b, Wbf, Ubf);
  attn_scores_kernel<<<dim3(YPAD / 16, B_), 512, 0, stream>>>(Ubf, xbf, alpha_out);
  attn_context_kernel<<<dim3(YPAD / 16, B_), 256, 0, stream>>>(alpha_out, xTbf,
                                                               final_w, final_b,
                                                               m_out, y_out);
  loss_kernel<<<1, 256, 0, stream>>>(y_out, target, loss_out);
}
